// MultiHeadSelfAttention_12730283065842
// MI455X (gfx1250) — matrix operand order repaired and checked by reading, not compiled, not run on hardware
//
#include <hip/hip_runtime.h>

// MHA for BS=4, S=2048, DIM=768, H=12, DH=64 on gfx1250 (MI455X).
// Pipeline: (1) fused QKV projection fp32->f16 (Q scaled, V transposed),
// (2) flash attention with v_wmma_f32_16x16x32_f16 + TDM tensor_load_to_lds
//     for K/V^T tile staging, (3) output projection.
// Workspace use: 3x QKV f16 (12.6MB each) + context f16 (12.6MB) = 50.4MB.

#define BSZ 4
#define SEQ 2048
#define DIMN 768
#define NH 12
#define DHD 64

typedef _Float16 half_t;
typedef __attribute__((ext_vector_type(16))) _Float16 v16h;
typedef __attribute__((ext_vector_type(8)))  _Float16 v8h;
typedef __attribute__((ext_vector_type(4)))  _Float16 v4h;
typedef __attribute__((ext_vector_type(8)))  float    v8f;
typedef __attribute__((ext_vector_type(4)))  unsigned int u32x4;
typedef __attribute__((ext_vector_type(8)))  int i32x8;
typedef __attribute__((ext_vector_type(4)))  int i32x4;

static __device__ __forceinline__ v16h cat16(v8h lo, v8h hi) {
  return __builtin_shufflevector(lo, hi, 0,1,2,3,4,5,6,7,8,9,10,11,12,13,14,15);
}
static __device__ __forceinline__ v8f wmma_f16(v16h a, v16h b, v8f c) {
  // v_wmma_f32_16x16x32_f16: D = A(16x32) * B(32x16) + C(16x16 f32)
  return __builtin_amdgcn_wmma_f32_16x16x32_f16(false, a, false, b, (short)0, c,
                                                false, false);
}
static __device__ __forceinline__ unsigned lds_off(const void* p) {
  // addrspace(3) pointer value == ds byte address (flat LDS aperture keeps it
  // in addr[31:0]); truncate the generic pointer.
  return (unsigned)(unsigned long long)p;
}

// Tensor Data Mover: 2D tile of 2-byte elements, global -> LDS, with LDS row
// padding.  D# built per CDNA5 ISA.
// tile_d0 = row length (elements), tile_d1 = #rows, pitch0 = memory row pitch
// (elements).  pad_interval/pad_amount encoded per D# group1 spec.
// This toolchain exposes the 6-arg builtin:
//   (uint32x4 g0, int32x8 g1, int32x4 g2, int32x4 g3, int32x8 g4, i32 cpol)
static __device__ __forceinline__ void tdm_load_2d_f16(
    const void* gaddr, unsigned ldsaddr, unsigned tile_d0, unsigned tile_d1,
    unsigned long long pitch0, unsigned pad_interval, unsigned pad_amount) {
  unsigned long long ga = (unsigned long long)gaddr;
  u32x4 g0;
  g0[0] = 1u;                                   // count=1, user descriptor
  g0[1] = ldsaddr;                              // lds_addr [63:32]
  g0[2] = (unsigned)(ga & 0xffffffffu);         // global_addr [95:64]
  g0[3] = (unsigned)((ga >> 32) & 0x01ffffffu)  // global_addr [120:96]
          | (2u << 30);                         // type=2 ("image") [127:126]

  unsigned s0lo = (unsigned)(pitch0 & 0xffffffffull);
  unsigned s0hi = (unsigned)(pitch0 >> 32) & 0xffffu;
  i32x8 g1;
  g1[0] = (int)((1u << 16)                       // data_size = 1 (2 bytes)
                | (1u << 20)                     // pad_enable
                | (pad_interval << 22)           // pad_interval [24:22]
                | (pad_amount << 25));           // pad_amount  [31:25]
  g1[1] = (int)((tile_d0 & 0xffffu) << 16);      // tensor_dim0[15:0] @ bits 63:48
  g1[2] = (int)(((tile_d0 >> 16) & 0xffffu)      // tensor_dim0[31:16]
                | ((tile_d1 & 0xffffu) << 16));  // tensor_dim1[15:0]
  g1[3] = (int)(((tile_d1 >> 16) & 0xffffu)      // tensor_dim1[31:16]
                | ((tile_d0 & 0xffffu) << 16));  // tile_dim0 @ bits 127:112
  g1[4] = (int)(tile_d1 & 0xffffu);              // tile_dim1; tile_dim2=0
  g1[5] = (int)s0lo;                             // tensor_dim0_stride[31:0]
  g1[6] = (int)s0hi;                             // stride[47:32]; dim1_stride lo=0
  g1[7] = 0;                                     // dim1_stride hi (unused, 2D)

  i32x4 gz4 = {};                                // groups 2/3: unused (2D tile)
  i32x8 gz8 = {};                                // trailing group: unused
  __builtin_amdgcn_tensor_load_to_lds(g0, g1, gz4, gz4, gz8, 0);
}

// ---------------------------------------------------------------------------
// Kernel 1: fused QKV projection.  C = X @ W^T + b, 128x128 tile per block,
// 8 waves each computing a 32x64 sub-tile (2x4 WMMA accumulators).
// WHICH = 0: Q (scaled 1/sqrt(DH), layout (b,h,s,dh))
// WHICH = 1: K (layout (b,h,s,dh))
// WHICH = 2: V (layout (b,h,dh,s)  == transposed for the PV stage)
// ---------------------------------------------------------------------------
template <int WHICH>
__global__ __launch_bounds__(256) void qkv_proj_kernel(
    const float* __restrict__ X, const float* __restrict__ W,
    const float* __restrict__ Bb, half_t* __restrict__ dst)
{
  constexpr int LDA = 48;  // 32 + 16 halves pad (96B row: 16B-aligned, bank-spread)
  __shared__ __align__(16) half_t As[128 * LDA];
  __shared__ __align__(16) half_t Bs[128 * LDA];

  const int mbase = blockIdx.y * 128;
  const int nbase = blockIdx.x * 128;
  const int tid  = threadIdx.x;
  const int lane = tid & 31;
  const int wv   = tid >> 5;       // 0..7
  const int mw   = wv >> 1;        // 0..3 : 32-row group
  const int nw   = wv & 1;         // 0..1 : 64-col group
  const int g    = lane >> 4;      // lane half-group
  const int l16  = lane & 15;

  v8f acc[2][4];
  for (int i = 0; i < 2; ++i)
    for (int j = 0; j < 4; ++j) acc[i][j] = {};

  for (int kt = 0; kt < DIMN; kt += 32) {
    __syncthreads();
    // Stage 128x32 fp32 tiles of X and W as f16 into LDS.
    for (int it = 0; it < 4; ++it) {
      int idx = tid + it * 256;          // 1024 float4 chunks
      int row = idx >> 3;
      int c4  = (idx & 7) << 2;
      float4 fa = *reinterpret_cast<const float4*>(
          &X[(size_t)(mbase + row) * DIMN + kt + c4]);
      v4h ha = {(half_t)fa.x, (half_t)fa.y, (half_t)fa.z, (half_t)fa.w};
      *reinterpret_cast<v4h*>(&As[row * LDA + c4]) = ha;
      float4 fb = *reinterpret_cast<const float4*>(
          &W[(size_t)(nbase + row) * DIMN + kt + c4]);
      v4h hb = {(half_t)fb.x, (half_t)fb.y, (half_t)fb.z, (half_t)fb.w};
      *reinterpret_cast<v4h*>(&Bs[row * LDA + c4]) = hb;
    }
    if (kt + 32 < DIMN) {  // prefetch next k-tiles (global_prefetch_b8)
      int prow = tid & 127;
      __builtin_prefetch(&X[(size_t)(mbase + prow) * DIMN + kt + 32], 0, 1);
      __builtin_prefetch(&W[(size_t)(nbase + prow) * DIMN + kt + 32], 0, 1);
    }
    __syncthreads();

    v16h afrag[2], bfrag[4];
    for (int am = 0; am < 2; ++am) {
      int row = mw * 32 + am * 16 + l16;   // A: K chunks g*8 and 16+g*8
      afrag[am] = cat16(*reinterpret_cast<const v8h*>(&As[row * LDA + g * 8]),
                        *reinterpret_cast<const v8h*>(&As[row * LDA + 16 + g * 8]));
    }
    for (int bn = 0; bn < 4; ++bn) {
      int row = nw * 64 + bn * 16 + l16;   // B: K chunks g*8 and 16+g*8
      bfrag[bn] = cat16(*reinterpret_cast<const v8h*>(&Bs[row * LDA + g * 8]),
                        *reinterpret_cast<const v8h*>(&Bs[row * LDA + 16 + g * 8]));
    }
    for (int am = 0; am < 2; ++am)
      for (int bn = 0; bn < 4; ++bn)
        acc[am][bn] = wmma_f16(afrag[am], bfrag[bn], acc[am][bn]);
  }

  // Epilogue: bias add + layout transform + f16 store (branch-free: WHICH is
  // a template parameter).
  for (int am = 0; am < 2; ++am) {
    for (int bn = 0; bn < 4; ++bn) {
      int n = nbase + nw * 64 + bn * 16 + l16;
      float bias = Bb[n];
      int hh = n >> 6, dh = n & 63;
      for (int r = 0; r < 8; ++r) {
        int m  = mbase + mw * 32 + am * 16 + r + 8 * g;
        int bb = m >> 11, ss = m & (SEQ - 1);
        float v = acc[am][bn][r] + bias;
        size_t bh = (size_t)bb * NH + hh;
        if constexpr (WHICH == 0)
          dst[(bh * SEQ + ss) * DHD + dh] = (half_t)(v * 0.125f);  // 1/sqrt(64)
        else if constexpr (WHICH == 1)
          dst[(bh * SEQ + ss) * DHD + dh] = (half_t)v;
        else
          dst[(bh * DHD + dh) * SEQ + ss] = (half_t)v;
      }
    }
  }
}

// ---------------------------------------------------------------------------
// Kernel 2: flash attention.  Block = 8 waves x 16 q-rows = 128 q-rows per
// (batch, head, q-tile).  K/V^T tiles DMA'd into LDS by the Tensor Data Mover
// (wave 0 issues, s_wait_tensorcnt + barrier publishes); online softmax with
// wave32 shfl_xor reductions; P tile round-trips per-wave through LDS to
// convert C-layout -> A-layout (ordered with s_wait_dscnt).
// ---------------------------------------------------------------------------
__global__ __launch_bounds__(256) void attn_kernel(
    const half_t* __restrict__ qf, const half_t* __restrict__ kf,
    const half_t* __restrict__ vtf, const int* __restrict__ mask,
    half_t* __restrict__ ctx)
{
  constexpr int LDK = 72;    // 64 + 8 halves pad  (TDM: pad 16B per 128B row)
  constexpr int LDV = 136;   // 128 + 8 halves pad (TDM: pad 16B per 256B row)
  __shared__ __align__(16) half_t Ks[128 * LDK];       // K tile, row-major (key, dh)
  __shared__ __align__(16) half_t Vts[64 * LDV];       // V^T tile (dh, key)
  __shared__ __align__(16) half_t Ps[8 * 16 * LDV];    // per-wave P (16 x 128)
  __shared__ float mb[128];                            // mask bias per key

  const int qt = blockIdx.x, hh = blockIdx.y, bb = blockIdx.z;
  const int tid = threadIdx.x, lane = tid & 31, wv = tid >> 5;
  const int g = lane >> 4, l16 = lane & 15;

  const size_t bh = (size_t)bb * NH + hh;
  const unsigned ks_lds = lds_off(Ks);
  const unsigned vt_lds = lds_off(Vts);

  // Q A-fragments straight from global f16 (two 16B chunks per lane per frag).
  v16h aq[2];
  {
    const half_t* qrow =
        qf + (bh * SEQ + (size_t)qt * 128 + wv * 16 + l16) * DHD;
    for (int k2 = 0; k2 < 2; ++k2)
      aq[k2] = cat16(*reinterpret_cast<const v8h*>(qrow + k2 * 32 + g * 8),
                     *reinterpret_cast<const v8h*>(qrow + k2 * 32 + 16 + g * 8));
  }

  float rowmax[8], rowsum[8];
  for (int r = 0; r < 8; ++r) { rowmax[r] = -3.0e38f; rowsum[r] = 0.0f; }
  v8f acc[4];
  for (int d = 0; d < 4; ++d) acc[d] = {};

  half_t* pw = &Ps[wv * 16 * LDV];

  for (int kt = 0; kt < SEQ; kt += 128) {
    __syncthreads();   // prior-iteration LDS consumers done
    if (wv == 0) {
      // K tile: 128 rows x 64 f16, pitch 64; LDS pad 4 DWORDs per 32 DWORDs.
      tdm_load_2d_f16(kf + (bh * SEQ + kt) * DHD, ks_lds,
                      /*tile_d0=*/DHD, /*tile_d1=*/128,
                      /*pitch0=*/DHD, /*pad_interval=*/4, /*pad_amount=*/3);
      // V^T tile: 64 rows x 128 f16, pitch SEQ; pad 4 DWORDs per 64 DWORDs.
      tdm_load_2d_f16(vtf + bh * DHD * SEQ + kt, vt_lds,
                      /*tile_d0=*/128, /*tile_d1=*/DHD,
                      /*pitch0=*/SEQ, /*pad_interval=*/5, /*pad_amount=*/3);
    }
    if (tid < 128) mb[tid] = mask[bb * SEQ + kt + tid] ? 0.0f : -1.0e30f;
    if (wv == 0) __builtin_amdgcn_s_wait_tensorcnt(0);
    __syncthreads();   // publish TDM-filled tiles to all waves

    // Scores: 16x128 per wave = 8 frags, 2 WMMAs each (DH=64 -> k=2x32).
    v8f sfrag[8];
    for (int nf = 0; nf < 8; ++nf) {
      v8f c = {};
      int row = nf * 16 + l16;
      for (int k2 = 0; k2 < 2; ++k2) {
        v16h bk = cat16(
            *reinterpret_cast<const v8h*>(&Ks[row * LDK + k2 * 32 + g * 8]),
            *reinterpret_cast<const v8h*>(&Ks[row * LDK + k2 * 32 + 16 + g * 8]));
        c = wmma_f16(aq[k2], bk, c);
      }
      float bias = mb[nf * 16 + l16];     // bias depends on key column only
      for (int r = 0; r < 8; ++r) c[r] += bias;
      sfrag[nf] = c;
    }

    // Online softmax: tile row-max (reduce across 16 lanes carrying one row).
    float tmax[8];
    for (int r = 0; r < 8; ++r) {
      float v = sfrag[0][r];
      for (int nf = 1; nf < 8; ++nf) v = fmaxf(v, sfrag[nf][r]);
      for (int mM = 8; mM >= 1; mM >>= 1) v = fmaxf(v, __shfl_xor(v, mM, 32));
      tmax[r] = v;
    }
    float scale[8];
    for (int r = 0; r < 8; ++r) {
      float nm = fmaxf(rowmax[r], tmax[r]);
      scale[r] = __expf(rowmax[r] - nm);
      rowmax[r] = nm;
    }
    float tsum[8];
    for (int r = 0; r < 8; ++r) tsum[r] = 0.0f;
    for (int nf = 0; nf < 8; ++nf)
      for (int r = 0; r < 8; ++r) {
        float p = __expf(sfrag[nf][r] - rowmax[r]);
        sfrag[nf][r] = p;
        tsum[r] += p;
      }
    for (int r = 0; r < 8; ++r) {
      float v = tsum[r];
      for (int mM = 8; mM >= 1; mM >>= 1) v += __shfl_xor(v, mM, 32);
      rowsum[r] = rowsum[r] * scale[r] + v;
    }
    for (int d = 0; d < 4; ++d)
      for (int r = 0; r < 8; ++r) acc[d][r] *= scale[r];

    // P: C-layout regs -> per-wave LDS tile (row m = r + 8*g, col = key).
    for (int nf = 0; nf < 8; ++nf)
      for (int r = 0; r < 8; ++r)
        pw[(r + 8 * g) * LDV + nf * 16 + l16] = (half_t)sfrag[nf][r];
    asm volatile("s_wait_dscnt 0" ::: "memory");  // intra-wave LDS RAW fence

    // PV: context(16x64) += P(16x128) @ V(128x64), 4 k-steps x 4 d-frags.
    for (int ks = 0; ks < 4; ++ks) {
      v16h ap = cat16(
          *reinterpret_cast<const v8h*>(&pw[l16 * LDV + ks * 32 + g * 8]),
          *reinterpret_cast<const v8h*>(&pw[l16 * LDV + ks * 32 + 16 + g * 8]));
      for (int d = 0; d < 4; ++d) {
        int vr = d * 16 + l16;
        v16h bv = cat16(
            *reinterpret_cast<const v8h*>(&Vts[vr * LDV + ks * 32 + g * 8]),
            *reinterpret_cast<const v8h*>(&Vts[vr * LDV + ks * 32 + 16 + g * 8]));
        acc[d] = wmma_f16(ap, bv, acc[d]);
      }
    }
  }

  // Normalize and store context f16 in (b, s, dim) layout.
  float inv[8];
  for (int r = 0; r < 8; ++r) inv[r] = 1.0f / rowsum[r];
  for (int d = 0; d < 4; ++d)
    for (int r = 0; r < 8; ++r) {
      int m   = qt * 128 + wv * 16 + r + 8 * g;
      int col = hh * DHD + d * 16 + l16;
      ctx[((size_t)bb * SEQ + m) * DIMN + col] = (half_t)(acc[d][r] * inv[r]);
    }
}

// ---------------------------------------------------------------------------
// Kernel 3: output projection.  out = ctx(f16) @ o_w^T + o_b, fp32 result.
// ---------------------------------------------------------------------------
__global__ __launch_bounds__(256) void out_proj_kernel(
    const half_t* __restrict__ ctx, const float* __restrict__ o_w,
    const float* __restrict__ o_b, float* __restrict__ out)
{
  constexpr int LDA = 48;
  __shared__ __align__(16) half_t As[128 * LDA];
  __shared__ __align__(16) half_t Bs[128 * LDA];

  const int mbase = blockIdx.y * 128;
  const int nbase = blockIdx.x * 128;
  const int tid = threadIdx.x, lane = tid & 31, wv = tid >> 5;
  const int mw = wv >> 1, nw = wv & 1;
  const int g = lane >> 4, l16 = lane & 15;

  v8f acc[2][4];
  for (int i = 0; i < 2; ++i)
    for (int j = 0; j < 4; ++j) acc[i][j] = {};

  for (int kt = 0; kt < DIMN; kt += 32) {
    __syncthreads();
    for (int it = 0; it < 2; ++it) {       // A tile already f16: 512 x 16B
      int idx = tid + it * 256;
      int row = idx >> 2;
      int c8  = (idx & 3) << 3;
      *reinterpret_cast<v8h*>(&As[row * LDA + c8]) =
          *reinterpret_cast<const v8h*>(&ctx[(size_t)(mbase + row) * DIMN + kt + c8]);
    }
    for (int it = 0; it < 4; ++it) {       // B tile fp32 -> f16
      int idx = tid + it * 256;
      int row = idx >> 3;
      int c4  = (idx & 7) << 2;
      float4 fb = *reinterpret_cast<const float4*>(
          &o_w[(size_t)(nbase + row) * DIMN + kt + c4]);
      v4h hb = {(half_t)fb.x, (half_t)fb.y, (half_t)fb.z, (half_t)fb.w};
      *reinterpret_cast<v4h*>(&Bs[row * LDA + c4]) = hb;
    }
    if (kt + 32 < DIMN) {
      int prow = tid & 127;
      __builtin_prefetch(&ctx[(size_t)(mbase + prow) * DIMN + kt + 32], 0, 1);
      __builtin_prefetch(&o_w[(size_t)(nbase + prow) * DIMN + kt + 32], 0, 1);
    }
    __syncthreads();

    v16h afrag[2], bfrag[4];
    for (int am = 0; am < 2; ++am) {
      int row = mw * 32 + am * 16 + l16;
      afrag[am] = cat16(*reinterpret_cast<const v8h*>(&As[row * LDA + g * 8]),
                        *reinterpret_cast<const v8h*>(&As[row * LDA + 16 + g * 8]));
    }
    for (int bn = 0; bn < 4; ++bn) {
      int row = nw * 64 + bn * 16 + l16;
      bfrag[bn] = cat16(*reinterpret_cast<const v8h*>(&Bs[row * LDA + g * 8]),
                        *reinterpret_cast<const v8h*>(&Bs[row * LDA + 16 + g * 8]));
    }
    for (int am = 0; am < 2; ++am)
      for (int bn = 0; bn < 4; ++bn)
        acc[am][bn] = wmma_f16(afrag[am], bfrag[bn], acc[am][bn]);
  }

  for (int am = 0; am < 2; ++am) {
    for (int bn = 0; bn < 4; ++bn) {
      int n = nbase + nw * 64 + bn * 16 + l16;
      float bias = o_b[n];
      for (int r = 0; r < 8; ++r) {
        int m = mbase + mw * 32 + am * 16 + r + 8 * g;
        out[(size_t)m * DIMN + n] = acc[am][bn][r] + bias;
      }
    }
  }
}

// ---------------------------------------------------------------------------
extern "C" void kernel_launch(void* const* d_in, const int* in_sizes, int n_in,
                              void* d_out, int out_size, void* d_ws, size_t ws_size,
                              hipStream_t stream) {
  (void)in_sizes; (void)n_in; (void)out_size; (void)ws_size;
  const float* query = (const float*)d_in[0];
  const float* key_t = (const float*)d_in[1];
  const float* value = (const float*)d_in[2];
  const int*   mask  = (const int*)d_in[3];
  const float* q_w = (const float*)d_in[4];
  const float* q_b = (const float*)d_in[5];
  const float* k_w = (const float*)d_in[6];
  const float* k_b = (const float*)d_in[7];
  const float* v_w = (const float*)d_in[8];
  const float* v_b = (const float*)d_in[9];
  const float* o_w = (const float*)d_in[10];
  const float* o_b = (const float*)d_in[11];
  float* out = (float*)d_out;

  // Workspace layout (f16): Q | K | V^T | context = 4 x 12.58 MB = 50.4 MB.
  const size_t elems = (size_t)BSZ * NH * SEQ * DHD;  // == BSZ*SEQ*DIMN
  half_t* qf  = (half_t*)d_ws;
  half_t* kfp = qf  + elems;
  half_t* vtf = kfp + elems;
  half_t* ctx = vtf + elems;

  dim3 blk(256);
  dim3 gproj(DIMN / 128, (BSZ * SEQ) / 128);
  qkv_proj_kernel<0><<<gproj, blk, 0, stream>>>(query, q_w, q_b, qf);
  qkv_proj_kernel<1><<<gproj, blk, 0, stream>>>(key_t, k_w, k_b, kfp);
  qkv_proj_kernel<2><<<gproj, blk, 0, stream>>>(value, v_w, v_b, vtf);
  attn_kernel<<<dim3(SEQ / 128, NH, BSZ), blk, 0, stream>>>(
      qf, kfp, vtf, mask, ctx);
  out_proj_kernel<<<dim3(DIMN / 128, (BSZ * SEQ) / 128), blk, 0, stream>>>(
      ctx, o_w, o_b, out);
}
